// MixingAsLayer_24936580121078
// MI455X (gfx1250) — matrix operand order repaired and checked by reading, not compiled, not run on hardware
//
#include <hip/hip_runtime.h>
#include <hip/hip_bf16.h>
#include <math.h>

// ---------------------------------------------------------------------------
// MixingAsLayer fused CDNA5 kernel, v4.
//
// GEMM core:  [T=8192 x J=12288] @ [J x N=128]  in bf16 WMMA, f32 accum.
//   A[t, j=(l,f,g)] = sum_{s in l} y[t,f,s]*xg[t,g,s]   (built on the fly)
//   B[j, n] = n<64 ? wx0[n,f,g,l] : wy0[n-64,f,g,l]      (pre-swizzled bf16)
//
// Tiling: block = 64 tokens (4 WMMA M-subtiles), 512 threads = 16 waves:
//   wave = (nt = w&7 : 16-wide N-tile, kg = w>>3 : k-half).
// B fragments fetched from L2 once per block, reused x4 in registers.
// Per k-step all 4 subtile A-fragments are clause-loaded into distinct
// registers before the 4 WMMAs -> each WMMA waits only on its own pair.
// ---------------------------------------------------------------------------

typedef __attribute__((ext_vector_type(16))) __bf16 v16bf;
typedef __attribute__((ext_vector_type(8)))  __bf16 v8bf;
typedef __attribute__((ext_vector_type(2)))  __bf16 v2bf;
typedef __attribute__((ext_vector_type(8)))  float  v8f;
typedef __attribute__((ext_vector_type(4)))  int    v4i;

union ABFrag {
    v16bf v;
    v8bf  h[2];
    v4i   i4[2];
};

#define TOK 64          // tokens per workgroup (4 WMMA M-subtiles)
#define JDIM 12288      // 3*64*64 contraction length
#define PLANE 4096      // bf16 elems per s-plane: [s][t][c], 64x64
#define CH_STRIDE 72    // bf16 row stride of a P chunk (16B aligned, 0-conflict)

// LDS layout (bytes); dynamic LDS total = 286720
#define OFF_XG    0        // __bf16 [9][64][64]   73728 B  (s-major planes)
#define OFF_Y     73728    // __bf16 [9][64][64]   73728 B
#define OFF_CHUNK 147456   // __bf16 [8][64][72]   73728 B
#define OFF_MIX0  221184   // float  [64][128]     32768 B
#define OFF_MIX1  253952   // float  [64][128]     32768 B
#define OFF_MXF   0        // float  [64][64][3]   (overlays dead xg region)
#define OFF_MYF   49152    // float  [64][64][3]   (overlays dead xg/y region)
#define LDS_BYTES 286720

// ---------------------------------------------------------------------------
// Prep: swizzle wx0/wy0 into bf16 WMMA-B fragments. Fragment = 1024 B,
// fragIdx = ks*8 + nt. Lane L holds column n = nt*16 + L%16 and 16
// consecutive k of the 32-step (lanes 0-15: k 0..15; 16-31: k 16..31).
// ---------------------------------------------------------------------------
__global__ __launch_bounds__(256) void prep_weights_kernel(
        const float* __restrict__ wx0, const float* __restrict__ wy0,
        __bf16* __restrict__ wfrag) {
    int e       = blockIdx.x * 256 + threadIdx.x;
    int fragIdx = e >> 9;
    int within  = e & 511;
    int L       = within >> 4;
    int idx     = within & 15;
    int ks      = fragIdx >> 3;
    int nt      = fragIdx & 7;
    int j       = ks * 32 + ((L < 16) ? 0 : 16) + idx;
    int n       = nt * 16 + (L & 15);
    int l       = j >> 12;
    int f       = (j >> 6) & 63;
    int g       = j & 63;
    float v = (n < 64) ? wx0[((n * 64 + f) * 64 + g) * 3 + l]
                       : wy0[(((n - 64) * 64 + f) * 64 + g) * 3 + l];
    wfrag[e] = (__bf16)v;
}

// ---------------------------------------------------------------------------
// Fused main kernel: 128 blocks x 512 threads (16 wave32).
// ---------------------------------------------------------------------------
__global__ __launch_bounds__(512) void mixing_main_kernel(
        const float* __restrict__ x, const float* __restrict__ y,
        const __bf16* __restrict__ wfrag,
        const float* __restrict__ wx_h, const float* __restrict__ wy_h,
        const float* __restrict__ wxf,  const float* __restrict__ wyf,
        float* __restrict__ out) {
    extern __shared__ char lds[];
    __bf16* xgL   = (__bf16*)(lds + OFF_XG);
    __bf16* yL    = (__bf16*)(lds + OFF_Y);
    __bf16* chunk = (__bf16*)(lds + OFF_CHUNK);
    float*  mix0  = (float*)(lds + OFF_MIX0);
    float*  mix1  = (float*)(lds + OFF_MIX1);
    float*  mxfL  = (float*)(lds + OFF_MXF);
    float*  myfL  = (float*)(lds + OFF_MYF);

    const int  tid  = threadIdx.x;
    const int  lane = tid & 31;
    const int  wvu  = __builtin_amdgcn_readfirstlane(tid >> 5);  // 0..15
    const int  nt   = wvu & 7;      // N-tile
    const int  kg   = wvu >> 3;     // k-half
    const long tok0 = (long)blockIdx.x * TOK;

    // ---- Stage 1: load x,y; gate=(x1 x y1).y[0:3]; xg,y -> s-major planes
    for (int r = 0; r < 8; ++r) {
        int tf = tid + 512 * r;              // 64*64 = 4096 (t,f) pairs
        int t = tf >> 6, f = tf & 63;
        const float* xp = x + ((tok0 + t) * 64 + f) * 9;
        const float* yp = y + ((tok0 + t) * 64 + f) * 9;
        float xv[9], yv[9];
#pragma unroll
        for (int s = 0; s < 9; ++s) { xv[s] = xp[s]; yv[s] = yp[s]; }
        float c0 = xv[2] * yv[3] - xv[3] * yv[2];
        float c1 = xv[3] * yv[1] - xv[1] * yv[3];
        float c2 = xv[1] * yv[2] - xv[2] * yv[1];
        float gate = c0 * yv[0] + c1 * yv[1] + c2 * yv[2];
#pragma unroll
        for (int s = 0; s < 9; ++s) {
            xgL[s * PLANE + t * 64 + f] = (__bf16)(xv[s] * gate);
            yL [s * PLANE + t * 64 + f] = (__bf16)yv[s];
        }
    }
    __syncthreads();

    // ---- Stage 2: fused P-build + WMMA GEMM
    v8f acc[4];
#pragma unroll
    for (int st = 0; st < 4; ++st) acc[st] = (v8f){0.f,0.f,0.f,0.f,0.f,0.f,0.f,0.f};

    const int aoff0 = (lane < 16) ? 0 : 16;  // A-frag k-run start (contiguous k = 16h + s)
    const int arow  = lane & 15;             // A-frag row within subtile
    const int gp2   = lane * 2;              // g pair handled in build

    for (int si = 0; si < 24; ++si) {
        // Waves {c, c+8} build chunk ci = si*8+c: kg selects 32 t-rows.
        {
            int ci = __builtin_amdgcn_readfirstlane(si * 8 + nt);
            int l  = ci >> 6;                // uniform 0,1,2
            int f  = ci & 63;
            __bf16* cslot = chunk + nt * (TOK * CH_STRIDE);
#pragma unroll 4
            for (int i = 0; i < 32; ++i) {   // one t-row per iteration
                int t = kg * 32 + i;
                const __bf16* yb = yL  + t * 64 + f;   // + s*PLANE
                const __bf16* xb = xgL + t * 64 + gp2; // + s*PLANE
                float p0, p1;
                switch (l) {                 // specialized s-range: l*l..+2l
                case 0: {
                    float ys = (float)yb[0];
                    v2bf xp = *(const v2bf*)(xb);
                    p0 = ys * (float)xp[0]; p1 = ys * (float)xp[1];
                    break; }
                case 1: {
                    p0 = 0.f; p1 = 0.f;
#pragma unroll
                    for (int s = 1; s <= 3; ++s) {
                        float ys = (float)yb[s * PLANE];
                        v2bf xp = *(const v2bf*)(xb + s * PLANE);
                        p0 += ys * (float)xp[0]; p1 += ys * (float)xp[1];
                    }
                    break; }
                default: {
                    p0 = 0.f; p1 = 0.f;
#pragma unroll
                    for (int s = 4; s <= 8; ++s) {
                        float ys = (float)yb[s * PLANE];
                        v2bf xp = *(const v2bf*)(xb + s * PLANE);
                        p0 += ys * (float)xp[0]; p1 += ys * (float)xp[1];
                    }
                    break; }
                }
                v2bf pv = {(__bf16)p0, (__bf16)p1};
                *(v2bf*)(cslot + t * CH_STRIDE + gp2) = pv;
            }
        }
        __syncthreads();

        // Batch-load this wave's 8 B fragments (k-half kg), then 32 WMMAs.
        ABFrag B[8];
#pragma unroll
        for (int q = 0; q < 8; ++q) {        // q = (c2 - kg*4)*2 + kk
            int ks = (si * 8 + kg * 4 + (q >> 1)) * 2 + (q & 1);
            const char* bp = (const char*)wfrag
                           + ((size_t)(ks * 8 + nt)) * 1024 + lane * 32;
            B[q].i4[0] = *(const v4i*)(bp);
            B[q].i4[1] = *(const v4i*)(bp + 16);
            __builtin_prefetch(bp + 131072, 0, 3);  // next round (+16 ks)
        }
#pragma unroll
        for (int q = 0; q < 8; ++q) {
            int c2 = kg * 4 + (q >> 1);
            int kk = q & 1;
            const __bf16* cbase = chunk + c2 * (TOK * CH_STRIDE)
                                + kk * 32 + aoff0 + arow * CH_STRIDE;
            // Clause-load all 4 subtile A-fragments into distinct registers,
            // then issue the 4 WMMAs: each waits only on its own pair.
            ABFrag A0, A1, A2, A3;
            A0.h[0] = *(const v8bf*)(cbase);
            A0.h[1] = *(const v8bf*)(cbase + 8);
            A1.h[0] = *(const v8bf*)(cbase + 16 * CH_STRIDE);
            A1.h[1] = *(const v8bf*)(cbase + 16 * CH_STRIDE + 8);
            A2.h[0] = *(const v8bf*)(cbase + 32 * CH_STRIDE);
            A2.h[1] = *(const v8bf*)(cbase + 32 * CH_STRIDE + 8);
            A3.h[0] = *(const v8bf*)(cbase + 48 * CH_STRIDE);
            A3.h[1] = *(const v8bf*)(cbase + 48 * CH_STRIDE + 8);
            acc[0] = __builtin_amdgcn_wmma_f32_16x16x32_bf16(
                    false, A0.v, false, B[q].v, (short)0, acc[0], false, false);
            acc[1] = __builtin_amdgcn_wmma_f32_16x16x32_bf16(
                    false, A1.v, false, B[q].v, (short)0, acc[1], false, false);
            acc[2] = __builtin_amdgcn_wmma_f32_16x16x32_bf16(
                    false, A2.v, false, B[q].v, (short)0, acc[2], false, false);
            acc[3] = __builtin_amdgcn_wmma_f32_16x16x32_bf16(
                    false, A3.v, false, B[q].v, (short)0, acc[3], false, false);
        }
        __syncthreads();
    }

    // ---- Stage 3: combine k-halves (disjoint (t,n) per lane -> no atomics)
    {
        int t0 = (lane < 16) ? 0 : 8;
        int n  = (lane & 15) + 16 * nt;
        if (kg == 0) {
#pragma unroll
            for (int st = 0; st < 4; ++st)
#pragma unroll
                for (int r = 0; r < 8; ++r)
                    mix0[(st * 16 + t0 + r) * 128 + n] = acc[st][r];
        }
        __syncthreads();
        if (kg == 1) {
#pragma unroll
            for (int st = 0; st < 4; ++st)
#pragma unroll
                for (int r = 0; r < 8; ++r)
                    mix0[(st * 16 + t0 + r) * 128 + n] += acc[st][r];
        }
        __syncthreads();
    }

    // ---- Stage 4: two tanh MLP layers (64x64), x/y halves independent
    for (int layer = 0; layer < 2; ++layer) {
        const float* src = (layer & 1) ? mix1 : mix0;
        float*       dst = (layer & 1) ? mix0 : mix1;
        for (int r = 0; r < 16; ++r) {
            int o    = tid + 512 * r;        // 64*128 = 8192 outputs
            int t    = o >> 7;
            int n    = o & 127;
            int half = n >> 6;
            int k    = n & 63;
            const float* W  = (half ? wy_h : wx_h) + layer * 64 * 64;
            const float* sp = src + t * 128 + half * 64;
            float s = 0.f;
#pragma unroll 8
            for (int j = 0; j < 64; ++j) s += sp[j] * W[j * 64 + k];
            dst[t * 128 + half * 64 + k] = tanhf(s);
        }
        __syncthreads();
    }
    // Results back in mix0.  mxf/myf overlay the (dead) xg/y region.

    // ---- Stage 5: final projection tanh(mix @ w{x,y}f[l,f,:])
    for (int r = 0; r < 48; ++r) {
        int o    = tid + 512 * r;            // 2*64*64*3 = 24576 outputs
        int half = (o >= 12288) ? 1 : 0;
        int rem  = o - half * 12288;
        int t    = rem / 192;
        int r2   = rem - t * 192;
        int f    = r2 / 3;
        int l    = r2 - f * 3;
        const float* W  = (half ? wyf : wxf) + (l * 64 + f) * 64;
        const float* sp = mix0 + t * 128 + half * 64;
        float s = 0.f;
#pragma unroll 8
        for (int k = 0; k < 64; ++k) s += sp[k] * W[k];
        (half ? myfL : mxfL)[(t * 64 + f) * 3 + l] = tanhf(s);
    }
    __syncthreads();

    // ---- Stage 6: res = mxf[deg(s)]*xg + myf[deg(s)]*y
    // (x,y re-read fp32 from global; gate recomputed: xg region was overlaid)
    const int deg[9] = {0, 1, 1, 1, 2, 2, 2, 2, 2};
    for (int r = 0; r < 8; ++r) {
        int tf = tid + 512 * r;
        int t = tf >> 6, f = tf & 63;
        const float* xp = x + ((tok0 + t) * 64 + f) * 9;
        const float* yp = y + ((tok0 + t) * 64 + f) * 9;
        float xv[9], yv[9];
#pragma unroll
        for (int s = 0; s < 9; ++s) { xv[s] = xp[s]; yv[s] = yp[s]; }
        float c0 = xv[2] * yv[3] - xv[3] * yv[2];
        float c1 = xv[3] * yv[1] - xv[1] * yv[3];
        float c2 = xv[1] * yv[2] - xv[2] * yv[1];
        float gate = c0 * yv[0] + c1 * yv[1] + c2 * yv[2];
        const float* mx = mxfL + (t * 64 + f) * 3;
        const float* my = myfL + (t * 64 + f) * 3;
        float* op = out + ((tok0 + t) * 64 + f) * 9;
#pragma unroll
        for (int s = 0; s < 9; ++s)
            op[s] = mx[deg[s]] * (xv[s] * gate) + my[deg[s]] * yv[s];
    }
}

extern "C" void kernel_launch(void* const* d_in, const int* in_sizes, int n_in,
                              void* d_out, int out_size, void* d_ws, size_t ws_size,
                              hipStream_t stream) {
    (void)in_sizes; (void)n_in; (void)out_size; (void)ws_size;
    const float* x    = (const float*)d_in[0];
    const float* y    = (const float*)d_in[1];
    const float* wx0  = (const float*)d_in[2];
    const float* wy0  = (const float*)d_in[3];
    const float* wx_h = (const float*)d_in[4];
    const float* wy_h = (const float*)d_in[5];
    const float* wxf  = (const float*)d_in[6];
    const float* wyf  = (const float*)d_in[7];
    __bf16* wfrag = (__bf16*)d_ws;   // 12288*128 bf16 = 3,145,728 bytes

    prep_weights_kernel<<<(JDIM * 128) / 256, 256, 0, stream>>>(wx0, wy0, wfrag);

    // 8192 tokens / 64 per block = 128 blocks, 16 waves each
    mixing_main_kernel<<<128, 512, LDS_BYTES, stream>>>(
        x, y, wfrag, wx_h, wy_h, wxf, wyf, (float*)d_out);
}
